// TransformerPointNetSetAbstractionMsg3D_17540646436964
// MI455X (gfx1250) — compile-verified, matrix operand order checked by reading, not run on hardware
//
#include <hip/hip_runtime.h>
#include <hip/hip_bf16.h>

#define BQ 8
#define NP 4096
#define SP 1024

typedef __bf16 bf16;
typedef __bf16 v16bf __attribute__((ext_vector_type(16)));
typedef __bf16 v8bf  __attribute__((ext_vector_type(8)));
typedef float  v8f   __attribute__((ext_vector_type(8)));
typedef unsigned int u32x4 __attribute__((ext_vector_type(4)));
typedef int          i32x8 __attribute__((ext_vector_type(8)));
typedef int          i32x4 __attribute__((ext_vector_type(4)));

#if defined(__has_builtin)
#if __has_builtin(__builtin_amdgcn_tensor_load_to_lds) && __has_builtin(__builtin_amdgcn_s_wait_tensorcnt)
#define HAVE_TDM 1
#endif
#endif

__device__ __forceinline__ v8f wmma_bf16(v16bf a, v16bf b, v8f c) {
  return __builtin_amdgcn_wmma_f32_16x16x32_bf16(false, a, false, b, (short)0, c, false, false);
}

// Two contiguous 16-byte loads -> one 16x32 fragment register set.
__device__ __forceinline__ v16bf ld_frag(const bf16* p) {
  v8bf lo = *(const v8bf*)p;
  v8bf hi = *(const v8bf*)(p + 16);
  return __builtin_shufflevector(lo, hi, 0, 1, 2, 3, 4, 5, 6, 7, 8, 9, 10, 11, 12, 13, 14, 15);
}

// A fragment: 16x32 bf16, row-major source (ISA 7.12.2)
__device__ __forceinline__ v16bf load_a_frag(const bf16* m, int ld, int row0, int k0, int lane) {
  const int r  = row0 + (lane & 15);
  const int kb = (lane < 16) ? 0 : 8;
  return ld_frag(m + r * ld + k0 + kb);
}

// B fragment from B-transposed storage: element (k,n) at m[(col0+n)*ld + k0 + k]
__device__ __forceinline__ v16bf load_bt_frag(const bf16* m, int ld, int k0, int col0, int lane) {
  const int n  = col0 + (lane & 15);
  const int kb = (lane < 16) ? 0 : 8;
  return ld_frag(m + n * ld + k0 + kb);
}

// Block GEMM, 8 waves: C[M,N] = act(A[M,Kc](LDS) @ W + bias (+resid)).
// WT: global bf16, pre-transposed [N][Kc] with zero-padded K rows; B-fragments
// read it directly (resident in L2; two global_load_b128 per fragment).
template<int M, int N, int Kc, bool RELU, bool TRANSC>
__device__ __forceinline__ void gemm(const bf16* A, int lda,
                                     const bf16* WT,
                                     const float* bias,
                                     bf16* C, int ldc,
                                     const bf16* resid, int ldr,
                                     int lane, int wave) {
  constexpr int NT = N / 16;
  constexpr int NTILE = (M / 16) * NT;
  constexpr int TPW = (NTILE + 7) / 8;
  static_assert(TPW <= 4, "too many tiles per wave");
  __syncthreads();
  v8f acc[TPW];
  v8f z = {0.f, 0.f, 0.f, 0.f, 0.f, 0.f, 0.f, 0.f};
#pragma unroll
  for (int i = 0; i < TPW; ++i) acc[i] = z;
#pragma unroll
  for (int ti = 0; ti < TPW; ++ti) {
    int t = wave + ti * 8;
    if (t < NTILE) {
      int tm = t / NT, tn = t - tm * NT;
#pragma unroll
      for (int kc = 0; kc < Kc; kc += 32) {
        v16bf a = load_a_frag(A, lda, tm * 16, kc, lane);
        v16bf b = load_bt_frag(WT, Kc, kc, tn * 16, lane);
        acc[ti] = wmma_bf16(a, b, acc[ti]);
      }
    }
  }
#pragma unroll
  for (int ti = 0; ti < TPW; ++ti) {
    int t = wave + ti * 8;
    if (t < NTILE) {
      int tm = t / NT, tn = t - tm * NT;
      int n  = tn * 16 + (lane & 15);
      int mo = (lane < 16) ? 0 : 8;
      float bv = bias ? bias[n] : 0.f;
#pragma unroll
      for (int r = 0; r < 8; ++r) {
        int m = tm * 16 + mo + r;
        float v = acc[ti][r] + bv;
        if (resid) v += (float)resid[m * ldr + n];
        if (RELU) v = fmaxf(v, 0.f);
        if (TRANSC) C[n * ldc + m] = (bf16)v;
        else        C[m * ldc + n] = (bf16)v;
      }
    }
  }
  __syncthreads();
}

__device__ __forceinline__ void layernorm_rows(bf16* X, int ld, int M, int N,
                                               const float* g, const float* b, int tid) {
  if (tid < M) {
    float mean = 0.f;
    for (int c = 0; c < N; ++c) mean += (float)X[tid * ld + c];
    mean /= (float)N;
    float var = 0.f;
    for (int c = 0; c < N; ++c) {
      float d = (float)X[tid * ld + c] - mean;
      var += d * d;
    }
    var /= (float)N;
    float rs = rsqrtf(var + 1e-5f);
    for (int c = 0; c < N; ++c)
      X[tid * ld + c] = (bf16)(((float)X[tid * ld + c] - mean) * rs * g[c] + b[c]);
  }
  __syncthreads();
}

struct EncP {
  const bf16 *wqT, *wkT, *wvT, *woT, *w1T, *w2T;     // bf16, B-transposed, in d_ws
  const float *bq, *bk, *bv, *bo, *b1, *b2;
  const float *ln1g, *ln1b, *ln2g, *ln2b;
};
struct ScaleP {
  const bf16 *p0T, *p1T, *p2T;
  const float *pb0, *pb1, *pb2;
  EncP e[2];
};

// One transformer encoder layer over a K x 128 tile in XA (LDS bf16).
template<int K>
__device__ void encoder_layer(bf16* XA, bf16* XB, const EncP& p,
                              int tid, int lane, int wave) {
  constexpr int Kp = (K < 32) ? 32 : K;     // attention inner dim padded to 32
  bf16* O   = XB;                           // 64x128 (ld 128)
  bf16* Qh  = XB + 8192;                    // K x 32  (ld 32)
  bf16* Kh  = XB + 8192 + 2048;             // Kp x 32 (ld 32)
  bf16* Sc  = XB + 8192 + 4096;             // K x Kp  (ld Kp)
  bf16* VhT = Qh;                           // 32 x Kp (ld Kp), reuses Qh slot
  const float rsdh = 0.17677669529663687f;  // 1/sqrt(32)

  for (int h = 0; h < 4; ++h) {
    gemm<K, 32, 128, false, false>(XA, 128, p.wqT + h * 32 * 128, p.bq + h * 32,
                                   Qh, 32, nullptr, 0, lane, wave);
    gemm<K, 32, 128, false, false>(XA, 128, p.wkT + h * 32 * 128, p.bk + h * 32,
                                   Kh, 32, nullptr, 0, lane, wave);
    if (Kp > K) {  // zero K pad rows -> pad score columns come out 0
      for (int i = tid; i < (Kp - K) * 32; i += 256) Kh[K * 32 + i] = (bf16)0.f;
      __syncthreads();
    }
    // scores = (Qh @ Kh^T) * rsdh : Kh row-major [key][dh] == B transposed
    {
      constexpr int NT = Kp / 16;
      constexpr int NTILE = (K / 16) * NT;
#pragma unroll
      for (int ti = 0; ti < (NTILE + 7) / 8; ++ti) {
        int t = wave + ti * 8;
        if (t < NTILE) {
          int tm = t / NT, tn = t - tm * NT;
          v16bf a = load_a_frag(Qh, 32, tm * 16, 0, lane);
          v16bf b = load_bt_frag(Kh, 32, 0, tn * 16, lane);
          v8f zz = {0.f, 0.f, 0.f, 0.f, 0.f, 0.f, 0.f, 0.f};
          v8f c = wmma_bf16(a, b, zz);
          int n  = tn * 16 + (lane & 15);
          int mo = (lane < 16) ? 0 : 8;
#pragma unroll
          for (int r = 0; r < 8; ++r)
            Sc[(tm * 16 + mo + r) * Kp + n] = (bf16)(c[r] * rsdh);
        }
      }
      __syncthreads();
    }
    // row softmax over K valid columns (pad columns stay 0)
    if (tid < K) {
      float mx = -1e30f;
#pragma unroll 4
      for (int c = 0; c < K; ++c) mx = fmaxf(mx, (float)Sc[tid * Kp + c]);
      float sum = 0.f;
#pragma unroll 4
      for (int c = 0; c < K; ++c) sum += __expf((float)Sc[tid * Kp + c] - mx);
      float inv = 1.f / sum;
#pragma unroll 4
      for (int c = 0; c < K; ++c)
        Sc[tid * Kp + c] = (bf16)(__expf((float)Sc[tid * Kp + c] - mx) * inv);
    }
    __syncthreads();
    // V head written transposed (VhT[dh][key], ld=Kp): P@V B-loads contiguous
    gemm<K, 32, 128, false, true>(XA, 128, p.wvT + h * 32 * 128, p.bv + h * 32,
                                  VhT, Kp, nullptr, 0, lane, wave);
    if (Kp > K) {  // zero V pad keys
      for (int i = tid; i < (Kp - K) * 32; i += 256) {
        int m = K + i / 32, n = i - (i / 32) * 32;
        VhT[n * Kp + m] = (bf16)0.f;
      }
      __syncthreads();
    }
    // O[:, h*32 : +32] = softmax(scores) @ Vh
    {
      constexpr int NT = 2;
      constexpr int NTILE = (K / 16) * NT;
#pragma unroll
      for (int ti = 0; ti < (NTILE + 7) / 8; ++ti) {
        int t = wave + ti * 8;
        if (t < NTILE) {
          int tm = t / NT, tn = t - tm * NT;
          v8f acc = {0.f, 0.f, 0.f, 0.f, 0.f, 0.f, 0.f, 0.f};
#pragma unroll
          for (int kc = 0; kc < Kp; kc += 32) {
            v16bf a = load_a_frag(Sc, Kp, tm * 16, kc, lane);
            v16bf b = load_bt_frag(VhT, Kp, kc, tn * 16, lane);
            acc = wmma_bf16(a, b, acc);
          }
          int n  = tn * 16 + (lane & 15);
          int mo = (lane < 16) ? 0 : 8;
#pragma unroll
          for (int r = 0; r < 8; ++r)
            O[(tm * 16 + mo + r) * 128 + h * 32 + n] = (bf16)acc[r];
        }
      }
      __syncthreads();
    }
  }
  // x = LN1(x + O @ Wo + bo)
  gemm<K, 128, 128, false, false>(O, 128, p.woT, p.bo, XA, 128, XA, 128, lane, wave);
  layernorm_rows(XA, 128, K, 128, p.ln1g, p.ln1b, tid);
  // FFN: H = relu(x@W1+b1) (two 128-col halves), x = LN2(x + H@W2 + b2)
  gemm<K, 128, 128, true, false>(XA, 128, p.w1T, p.b1, XB, 256, nullptr, 0, lane, wave);
  gemm<K, 128, 128, true, false>(XA, 128, p.w1T + 128 * 128, p.b1 + 128, XB + 128, 256,
                                 nullptr, 0, lane, wave);
  gemm<K, 128, 256, false, false>(XB, 256, p.w2T, p.b2, XA, 128, XA, 128, lane, wave);
  layernorm_rows(XA, 128, K, 128, p.ln2g, p.ln2b, tid);
}

// ---------------- weight prep: fp32 [kreal x n] -> bf16 B-transposed [n x kpad] ----------------
struct PrepEnt { const float* src; bf16* dst; int kreal, n, kpad; };
struct PrepArgs { PrepEnt e[15]; };

__launch_bounds__(256)
__global__ void prep_kernel(PrepArgs pa) {
  PrepEnt E = pa.e[blockIdx.x];
  const int total = E.n * E.kpad;
  for (int i = threadIdx.x; i < total; i += 256) {
    int c = i / E.kpad, r = i - c * E.kpad;
    E.dst[i] = (r < E.kreal) ? (bf16)E.src[r * E.n + c] : (bf16)0.f;
  }
}

// ---------------- FPS: one 1024-thread block per batch ----------------
__launch_bounds__(1024)
__global__ void fps_kernel(const float* __restrict__ xyz, float* __restrict__ nxyz) {
  const int b = blockIdx.x, tid = threadIdx.x;
  __shared__ float sval[1024];
  __shared__ int   sidx[1024];
  __shared__ int   s_last;
  if (tid == 0) s_last = 0;
  float mind[4] = {1e10f, 1e10f, 1e10f, 1e10f};
  __syncthreads();
  for (int s = 0; s < SP; ++s) {
    const int last = s_last;
    const float* p = xyz + (b * NP + last) * 3;
    const float lx = p[0], ly = p[1], lz = p[2];
    if (tid == 0) {
      nxyz[(b * SP + s) * 3 + 0] = lx;
      nxyz[(b * SP + s) * 3 + 1] = ly;
      nxyz[(b * SP + s) * 3 + 2] = lz;
    }
    float bv = -1.f;
    int   bi = 0;
#pragma unroll
    for (int j = 0; j < 4; ++j) {
      int n = tid + j * 1024;
      const float* q = xyz + (b * NP + n) * 3;
      float dx = q[0] - lx, dy = q[1] - ly, dz = q[2] - lz;
      float d = dx * dx + dy * dy + dz * dz;
      mind[j] = fminf(mind[j], d);
      if (mind[j] > bv) { bv = mind[j]; bi = n; }
    }
    sval[tid] = bv; sidx[tid] = bi;
    __syncthreads();
    for (int off = 512; off > 0; off >>= 1) {
      if (tid < off && sval[tid + off] > sval[tid]) {
        sval[tid] = sval[tid + off];
        sidx[tid] = sidx[tid + off];
      }
      __syncthreads();
    }
    if (tid == 0) s_last = sidx[0];
    __syncthreads();
  }
}

#if HAVE_TDM
// TDM: 1-D tile of `nelem` dwords, global -> LDS (ISA 8.3/8.4 descriptor layout).
// 6-arg builtin form (clang-23 / therock headers): (u32x4, i32x8, i32x4, i32x4, i32x8, i32 cpol)
__device__ __forceinline__ void tdm_load_1d(unsigned lds_off, const void* gaddr, int nelem) {
  unsigned long long ga = (unsigned long long)gaddr;
  u32x4 g0 = {0u, 0u, 0u, 0u};
  g0[0] = 1u;                                            // count=1 valid user D#
  g0[1] = lds_off;                                       // lds_addr [63:32]
  g0[2] = (unsigned)(ga & 0xFFFFFFFFull);                // global_addr lo
  g0[3] = (unsigned)((ga >> 32) & 0x01FFFFFFull) | (2u << 30);  // addr[56:32] | type=2
  i32x8 g1 = {0, 0, 0, 0, 0, 0, 0, 0};
  g1[0] = (2 << 16);                                     // data_size = 4B
  g1[1] = (nelem & 0xFFFF) << 16;                        // tensor_dim0 lo16 @ [63:48]
  g1[2] = ((nelem >> 16) & 0xFFFF) | (1 << 16);          // tensor_dim0 hi16 | tensor_dim1=1
  g1[3] = (nelem & 0xFFFF) << 16;                        // tile_dim0 @ [127:112]
  g1[5] = nelem;                                         // tensor_dim0_stride lo32
  i32x4 gz4 = {0, 0, 0, 0};
  i32x8 gz8 = {0, 0, 0, 0, 0, 0, 0, 0};
  __builtin_amdgcn_tensor_load_to_lds(g0, g1, gz4, gz4, gz8, 0);
}
#endif

// ---------------- Ball query: thread-per-center, xyz tiles double-buffered via TDM ----------------
template<int K>
__launch_bounds__(256)
__global__ void ballq_kernel(const float* __restrict__ xyz, const float* __restrict__ nxyz,
                             int* __restrict__ ballidx, float r2) {
  __shared__ alignas(16) float P[2][256 * 3];
  const int tid = threadIdx.x;
  const int b = blockIdx.x / (SP / 256);
  const int s = (blockIdx.x % (SP / 256)) * 256 + tid;
  const int g = b * SP + s;
  const float cx = nxyz[g * 3 + 0], cy = nxyz[g * 3 + 1], cz = nxyz[g * 3 + 2];
  const float* src = xyz + b * NP * 3;
  int cnt = 0;
#if HAVE_TDM
  if (tid < 32) {  // wave 0 drives the tensor DMA
    tdm_load_1d((unsigned)(unsigned long long)&P[0][0], src, 768);
    __builtin_amdgcn_s_wait_tensorcnt(0);
  }
  __syncthreads();
  for (int t = 0; t < NP / 256; ++t) {
    const int cur = t & 1;
    if (tid < 32 && t + 1 < NP / 256)
      tdm_load_1d((unsigned)(unsigned long long)&P[cur ^ 1][0], src + (t + 1) * 768, 768);
#pragma unroll 4
    for (int j = 0; j < 256; ++j) {
      float dx = P[cur][j * 3 + 0] - cx, dy = P[cur][j * 3 + 1] - cy, dz = P[cur][j * 3 + 2] - cz;
      float d2 = dx * dx + dy * dy + dz * dz;
      if (d2 < r2 && cnt < K) { ballidx[g * K + cnt] = t * 256 + j; ++cnt; }
    }
    if (tid < 32) __builtin_amdgcn_s_wait_tensorcnt(0);
    __syncthreads();
  }
#else
  for (int t = 0; t < NP / 256; ++t) {
    __syncthreads();
    const float* q = src + (t * 256 + tid) * 3;
    P[0][tid * 3 + 0] = q[0];
    P[0][tid * 3 + 1] = q[1];
    P[0][tid * 3 + 2] = q[2];
    __syncthreads();
#pragma unroll 4
    for (int j = 0; j < 256; ++j) {
      float dx = P[0][j * 3 + 0] - cx, dy = P[0][j * 3 + 1] - cy, dz = P[0][j * 3 + 2] - cz;
      float d2 = dx * dx + dy * dy + dz * dz;
      if (d2 < r2 && cnt < K) { ballidx[g * K + cnt] = t * 256 + j; ++cnt; }
    }
  }
#endif
  const int pad = (cnt > 0) ? ballidx[g * K] : 0;
  for (int k = cnt; k < K; ++k) ballidx[g * K + k] = pad;
}

// ---------------- Fused per-group: gather -> MLP -> 2x encoder -> maxpool ----------------
template<int K, int C0, int C1>
__launch_bounds__(256)
__global__ void scale_kernel(const float* __restrict__ xyz, const float* __restrict__ feats,
                             const float* __restrict__ nxyz, const int* __restrict__ ballidx,
                             ScaleP sp, float* __restrict__ outf, int si) {
  __shared__ alignas(16) bf16 XA[64 * 128];   // activations (16 KB)
  __shared__ alignas(16) bf16 XB[64 * 256];   // scratch (32 KB)
  const int g = blockIdx.x;
  const int b = g / SP;
  const int tid = threadIdx.x, lane = tid & 31, wave = (tid >> 5) & 7;

  {
    v8bf z = {};
    for (int i = tid; i < 64 * 16; i += 256) ((v8bf*)XA)[i] = z;
  }
  __syncthreads();

  // gather: rel-xyz (3) + features (64), zero padded to 96 channels
  const float cx = nxyz[g * 3 + 0], cy = nxyz[g * 3 + 1], cz = nxyz[g * 3 + 2];
  for (int row = wave; row < K; row += 8) {
    int pi = ballidx[g * K + row];
    const float* px = xyz + (b * NP + pi) * 3;
    const float* pf = feats + (b * NP + pi) * 64;
    if (lane == 0) {
      XA[row * 128 + 0] = (bf16)(px[0] - cx);
      XA[row * 128 + 1] = (bf16)(px[1] - cy);
      XA[row * 128 + 2] = (bf16)(px[2] - cz);
    }
#pragma unroll 2
    for (int c = lane; c < 64; c += 32)
      XA[row * 128 + 3 + c] = (bf16)pf[c];
  }

  // PointNet MLP stack (bf16 WMMA, fp32 accumulate, ReLU)
  gemm<K, C0, 96, true, false>(XA, 128, sp.p0T, sp.pb0, XB, 128, nullptr, 0, lane, wave);
  gemm<K, C1, C0, true, false>(XB, 128, sp.p1T, sp.pb1, XA, 128, nullptr, 0, lane, wave);
  gemm<K, 128, C1, true, false>(XA, 128, sp.p2T, sp.pb2, XB, 128, nullptr, 0, lane, wave);
  for (int i = tid; i < K * 16; i += 256) ((v8bf*)XA)[i] = ((const v8bf*)XB)[i];

  encoder_layer<K>(XA, XB, sp.e[0], tid, lane, wave);
  encoder_layer<K>(XA, XB, sp.e[1], tid, lane, wave);

  // max-pool over the K group members
  for (int c = tid; c < 128; c += 256) {
    float m = -1e30f;
#pragma unroll 4
    for (int r = 0; r < K; ++r) m = fmaxf(m, (float)XA[r * 128 + c]);
    outf[g * 384 + si * 128 + c] = m;
  }
}

extern "C" void kernel_launch(void* const* d_in, const int* in_sizes, int n_in,
                              void* d_out, int out_size, void* d_ws, size_t ws_size,
                              hipStream_t stream) {
  (void)in_sizes; (void)n_in; (void)out_size; (void)ws_size;
  const float* xyz   = (const float*)d_in[0];
  const float* feats = (const float*)d_in[1];
  auto F = [&](int i) { return (const float*)d_in[i]; };

  float* out  = (float*)d_out;
  float* nxyz = out;               // [B,S,3]
  float* outf = out + BQ * SP * 3; // [B,S,384]
  int*   ballidx = (int*)d_ws;     // [B,S,64] reused per scale (stream-serial)
  bf16*  wbase = (bf16*)((char*)d_ws + (size_t)BQ * SP * 64 * sizeof(int));
  size_t woff = 0;
  auto alloc = [&](size_t n) { bf16* p = wbase + woff; woff += n; return p; };

  // params pytree leaf order: per scale dict keys sorted ('enc' < 'pn');
  // per encoder layer keys sorted: ln1,ln2,w1,w2,wk,wo,wq,wv; _lin = (W,b); ln = (g,b)
  const int C0s[3] = {32, 64, 64}, C1s[3] = {64, 64, 96};
  ScaleP   sp[3];
  PrepArgs pa[3];
  for (int si = 0; si < 3; ++si) {
    const int base = 2 + si * 38;
    const int C0 = C0s[si], C1 = C1s[si];
    ScaleP& p = sp[si];
    PrepArgs& q = pa[si];
    auto addw = [&](int slot, const float* src, size_t n_elem, int kreal, int n, int kpad) {
      bf16* d = alloc(n_elem);
      q.e[slot].src = src; q.e[slot].dst = d;
      q.e[slot].kreal = kreal; q.e[slot].n = n; q.e[slot].kpad = kpad;
      return (const bf16*)d;
    };
    // PointNet MLP weights (leaves at base+32..37)
    p.pb0 = F(base + 33);
    p.pb1 = F(base + 35);
    p.pb2 = F(base + 37);
    p.p0T = addw(0, F(base + 32), (size_t)C0 * 96, 67, C0, 96);
    p.p1T = addw(1, F(base + 34), (size_t)C1 * C0, C0, C1, C0);
    p.p2T = addw(2, F(base + 36), (size_t)128 * C1, C1, 128, C1);
    for (int l = 0; l < 2; ++l) {
      const int eb = base + l * 16;
      const int m = 3 + l * 6;
      EncP& e = p.e[l];
      e.ln1g = F(eb + 0);  e.ln1b = F(eb + 1);
      e.ln2g = F(eb + 2);  e.ln2b = F(eb + 3);
      e.b1 = F(eb + 5);  e.b2 = F(eb + 7);
      e.bk = F(eb + 9);  e.bo = F(eb + 11);
      e.bq = F(eb + 13); e.bv = F(eb + 15);
      e.w1T = addw(m + 0, F(eb + 4),  (size_t)256 * 128, 128, 256, 128);
      e.w2T = addw(m + 1, F(eb + 6),  (size_t)128 * 256, 256, 128, 256);
      e.wkT = addw(m + 2, F(eb + 8),  (size_t)128 * 128, 128, 128, 128);
      e.woT = addw(m + 3, F(eb + 10), (size_t)128 * 128, 128, 128, 128);
      e.wqT = addw(m + 4, F(eb + 12), (size_t)128 * 128, 128, 128, 128);
      e.wvT = addw(m + 5, F(eb + 14), (size_t)128 * 128, 128, 128, 128);
    }
  }

  prep_kernel<<<15, 256, 0, stream>>>(pa[0]);
  prep_kernel<<<15, 256, 0, stream>>>(pa[1]);
  prep_kernel<<<15, 256, 0, stream>>>(pa[2]);

  fps_kernel<<<BQ, 1024, 0, stream>>>(xyz, nxyz);

  ballq_kernel<16><<<BQ * (SP / 256), 256, 0, stream>>>(xyz, nxyz, ballidx, 0.1f * 0.1f);
  scale_kernel<16, 32, 64><<<BQ * SP, 256, 0, stream>>>(xyz, feats, nxyz, ballidx, sp[0], outf, 0);

  ballq_kernel<32><<<BQ * (SP / 256), 256, 0, stream>>>(xyz, nxyz, ballidx, 0.2f * 0.2f);
  scale_kernel<32, 64, 64><<<BQ * SP, 256, 0, stream>>>(xyz, feats, nxyz, ballidx, sp[1], outf, 1);

  ballq_kernel<64><<<BQ * (SP / 256), 256, 0, stream>>>(xyz, nxyz, ballidx, 0.4f * 0.4f);
  scale_kernel<64, 64, 96><<<BQ * SP, 256, 0, stream>>>(xyz, feats, nxyz, ballidx, sp[2], outf, 2);
}
